// Attention_13073880449373
// MI455X (gfx1250) — matrix operand order repaired and checked by reading, not compiled, not run on hardware
//
#include <hip/hip_runtime.h>
#include <hip/hip_bf16.h>

// ---------------------------------------------------------------------------
// Shapes (fixed by the reference)
// ---------------------------------------------------------------------------
#define BATCH 8
#define SEQ   1024
#define CDIM  1024
#define NHEAD 16
#define HDIM  64            // CDIM / NHEAD
#define N_QKV 3072          // 3*CDIM
#define PSTR  1040          // LDS prob row stride in f16 (1024 + 16 pad)
#define KROW  72            // LDS K row stride in f16 (64 + 8 pad = 144 B)

typedef _Float16 v8h  __attribute__((ext_vector_type(8)));
typedef _Float16 v16h __attribute__((ext_vector_type(16)));
typedef float    v8f  __attribute__((ext_vector_type(8)));
typedef unsigned int v4u __attribute__((ext_vector_type(4)));
typedef int      v4i  __attribute__((ext_vector_type(4)));
typedef int      v8i  __attribute__((ext_vector_type(8)));

// ---------------------------------------------------------------------------
// WMMA operand loaders (wave32, V_WMMA_F32_16X16X32_F16 layouts)
//
// A (16x32, f16): lane row = lane&15. half = lane>>4.
//   elems 0..7  : K = half*16 + j       (16B contiguous)
//   elems 8..15 : K = half*16 + 8 + j   (16B contiguous, +16B from first)
// B (32x16, f16): lane col = lane&15.
//   elems 0..15 : K = half*16 + j       (32B contiguous)
// C/D (16x16, f32): lane col = lane&15; VGPR r -> row = half*8 + r.
// ---------------------------------------------------------------------------
__device__ __forceinline__ v16h cat16(v8h lo, v8h hi) {
  return __builtin_shufflevector(lo, hi, 0, 1, 2, 3, 4, 5, 6, 7,
                                         8, 9, 10, 11, 12, 13, 14, 15);
}

// rowptr points at element [row][k0] of a row-major matrix
__device__ __forceinline__ v16h load_a16(const _Float16* rowptr, int half) {
  v8h lo = *(const v8h*)(rowptr + half * 16);
  v8h hi = *(const v8h*)(rowptr + half * 16 + 8);
  return cat16(lo, hi);
}

// p points at element [col][k0 + half*16] of the (N x K) operand source
__device__ __forceinline__ v16h load_b16(const _Float16* p) {
  v8h lo = *(const v8h*)(p);
  v8h hi = *(const v8h*)(p + 8);
  return cat16(lo, hi);
}

__device__ __forceinline__ v8f wmma_f16(v16h a, v16h b, v8f c) {
  return __builtin_amdgcn_wmma_f32_16x16x32_f16(false, a, false, b,
                                                (short)0, c, false, false);
}

// ---------------------------------------------------------------------------
// fp32 -> fp16 convert (grid-stride)
// ---------------------------------------------------------------------------
__global__ void k_f32_to_f16(const float* __restrict__ in,
                             _Float16* __restrict__ out, int n) {
  for (int i = blockIdx.x * blockDim.x + threadIdx.x; i < n;
       i += blockDim.x * gridDim.x)
    out[i] = (_Float16)in[i];
}

// transpose (K x N) fp32 -> (N x K) fp16; coalesced reads
__global__ void k_transpose_f16(const float* __restrict__ in,
                                _Float16* __restrict__ out, int K, int N) {
  for (int i = blockIdx.x * blockDim.x + threadIdx.x; i < K * N;
       i += blockDim.x * gridDim.x) {
    int n = i % N, k = i / N;
    out[(size_t)n * K + k] = (_Float16)in[(size_t)k * N + n];
  }
}

// ---------------------------------------------------------------------------
// WMMA GEMM with bias:  out[M,N] = A[M,K](f16) * BT[N,K]^T(f16) + bias[N]
// One wave computes a 16x64 tile (A reused across 4 N-tiles).
// ---------------------------------------------------------------------------
template <typename OUT_T>
__global__ void k_wmma_gemm_bias(const _Float16* __restrict__ A,
                                 const _Float16* __restrict__ BT,
                                 const float* __restrict__ bias,
                                 OUT_T* __restrict__ out,
                                 int M, int N, int K) {
  const int lane = threadIdx.x & 31;
  const int wid  = (blockIdx.x * blockDim.x + threadIdx.x) >> 5;
  const int nb4  = N >> 6;                       // 64-wide N tiles
  const int mt   = wid / nb4;
  const int nt   = wid % nb4;
  if (mt * 16 >= M) return;

  const int m0   = mt * 16;
  const int n0   = nt * 64;
  const int half = lane >> 4;
  const int lrow = lane & 15;

  const _Float16* arow = A + (size_t)(m0 + lrow) * K;
  const _Float16* brow0 = BT + (size_t)(n0 + lrow) * K + half * 16;

  v8f acc0 = {}, acc1 = {}, acc2 = {}, acc3 = {};
  for (int k0 = 0; k0 < K; k0 += 32) {
    __builtin_prefetch(arow + k0 + 128, 0, 1);
    v16h a  = load_a16(arow + k0, half);
    v16h b0 = load_b16(brow0 + (size_t)0 * 16 * K + k0);
    v16h b1 = load_b16(brow0 + (size_t)1 * 16 * K + k0);
    v16h b2 = load_b16(brow0 + (size_t)2 * 16 * K + k0);
    v16h b3 = load_b16(brow0 + (size_t)3 * 16 * K + k0);
    acc0 = wmma_f16(a, b0, acc0);
    acc1 = wmma_f16(a, b1, acc1);
    acc2 = wmma_f16(a, b2, acc2);
    acc3 = wmma_f16(a, b3, acc3);
  }

  v8f accs[4] = {acc0, acc1, acc2, acc3};
#pragma unroll
  for (int t = 0; t < 4; ++t) {
    const int n = n0 + t * 16 + lrow;
    const float bb = bias[n];
#pragma unroll
    for (int r = 0; r < 8; ++r) {
      const int m = m0 + half * 8 + r;
      out[(size_t)m * N + n] = (OUT_T)(accs[t][r] + bb);
    }
  }
}

// ---------------------------------------------------------------------------
// RoPE + reshape:
//  qkv16 (B,S,3,H,D) f16 -> qh (B,H,S,D), kh (B,H,S,D), vT (B,H,D,S)  all f16
// ---------------------------------------------------------------------------
__global__ void k_rope_reshape(const _Float16* __restrict__ qkv,
                               _Float16* __restrict__ qh,
                               _Float16* __restrict__ kh,
                               _Float16* __restrict__ vT) {
  const int id = blockIdx.x * blockDim.x + threadIdx.x;  // B*H*S*D threads
  const int d = id & 63;
  const int s = (id >> 6) & (SEQ - 1);
  const int h = (id >> 16) & (NHEAD - 1);
  const int b = id >> 20;

  const size_t base = ((size_t)(b * SEQ + s) * 3) * (NHEAD * HDIM) + h * HDIM;
  const int dp = (d < 32) ? d + 32 : d - 32;

  const float qv = (float)qkv[base + d];
  const float qp = (float)qkv[base + dp];
  const float kv = (float)qkv[base + (NHEAD * HDIM) + d];
  const float kp = (float)qkv[base + (NHEAD * HDIM) + dp];
  const float vv = (float)qkv[base + 2 * (NHEAD * HDIM) + d];

  const int j = d & 31;
  const float ang = (float)s * __expf(-0.28782313662425573f * (float)j);
  float sn, cs;
  __sincosf(ang, &sn, &cs);
  const float rsn = (d < 32) ? -sn : sn;

  const size_t bh = (size_t)(b * NHEAD + h);
  const size_t sd = (bh * SEQ + s) * HDIM + d;
  qh[sd] = (_Float16)(qv * cs + qp * rsn);
  kh[sd] = (_Float16)(kv * cs + kp * rsn);
  vT[(bh * HDIM + d) * SEQ + s] = (_Float16)vv;
}

// ---------------------------------------------------------------------------
// TDM: DMA the whole K matrix of one head (1024x64 f16, 128 KB) into LDS,
// inserting a 16B pad every 128B row (pad_interval=4 -> 128B, pad_amount=3
// -> 4 DWORDs) so WMMA A-operand ds_load_b128s are bank-conflict free.
// Descriptor layout per CDNA5 ISA.
// ---------------------------------------------------------------------------
__device__ __forceinline__ void tdm_load_k_head(const _Float16* gsrc,
                                                _Float16* lds_dst) {
  const unsigned lds_off = (unsigned)(unsigned long long)lds_dst; // LDS byte offset
  const unsigned long long ga = (unsigned long long)gsrc;
  const unsigned n8 = (SEQ * HDIM * 2) / 8;  // 16384 8-byte elements

  v4u g0;
  g0[0] = 1u;                                   // count=1, user descriptor
  g0[1] = lds_off;                              // lds_addr (bytes)
  g0[2] = (unsigned)ga;                         // global_addr[31:0]
  g0[3] = (unsigned)((ga >> 32) & 0x1FFFFFFu) | (2u << 30);  // addr hi | type=2

  v8i g1;
  g1[0] = (int)((3u << 16)      // data_size = 8B
              | (1u << 20)      // pad_enable
              | (4u << 22)      // pad_interval: pad every 128B
              | (3u << 25));    // pad_amount: 4 DWORDs (16B)
  g1[1] = (int)((n8 & 0xFFFFu) << 16);          // tensor_dim0[15:0]
  g1[2] = (int)((n8 >> 16) | (1u << 16));       // tensor_dim0[31:16] | tensor_dim1=1
  g1[3] = (int)((n8 & 0xFFFFu) << 16);          // tile_dim0 = n8 (fits 16b)
  g1[4] = 0;                                    // tile_dim1=0, tile_dim2=0
  g1[5] = (int)n8;                              // tensor_dim0_stride lo
  g1[6] = 0;
  g1[7] = 0;

  v4i z = {0, 0, 0, 0};
#if __has_builtin(__builtin_amdgcn_tensor_load_to_lds)
#if defined(__clang_major__) && (__clang_major__ >= 23)
  v8i z8 = {0, 0, 0, 0, 0, 0, 0, 0};
  __builtin_amdgcn_tensor_load_to_lds(g0, g1, z, z, z8, 0);
#else
  __builtin_amdgcn_tensor_load_to_lds(g0, g1, z, z, 0);
#endif
  __builtin_amdgcn_s_wait_tensorcnt(0);
#else
  (void)g0; (void)g1; (void)z;
  // Fallback: cooperative vector copy with row padding (one wave).
  const int lane = threadIdx.x & 31;
  for (int r = lane; r < SEQ; r += 32) {
    const v8h* s = (const v8h*)(gsrc + (size_t)r * HDIM);
    v8h* d = (v8h*)(lds_dst + (size_t)r * KROW);
#pragma unroll
    for (int c = 0; c < 8; ++c) d[c] = s[c];
  }
#endif
}

// ---------------------------------------------------------------------------
// Attention: block = 4 waves sharing one (b,h); each wave owns 16 queries.
//  Stage 0: TDM DMA of head's K matrix into LDS (row-padded)
//  Pass A : S^T tiles = K_tile(16xD) x Q^T(Dx16) via WMMA -> running max
//  Pass B : recompute, exp, sum, store unnormalized f16 probs to LDS
//           in the exact B-operand layout (row q, contiguous k, padded)
//  Phase 3: out^T = V^T(DxS) x P^T(Sx16) via WMMA, scale by 1/sum,
//           packed 16B f16 stores into attn (B,S,C)
// ---------------------------------------------------------------------------
__global__ void k_attention(const _Float16* __restrict__ qh,
                            const _Float16* __restrict__ kh,
                            const _Float16* __restrict__ vT,
                            _Float16* __restrict__ attn) {
  extern __shared__ _Float16 smem[];
  _Float16* ksh = smem;                       // SEQ x KROW f16 (144 KB)
  _Float16* pbase = smem + SEQ * KROW;        // 4 x 16 x PSTR f16 (130 KB)

  const int lane = threadIdx.x & 31;
  const int warp = threadIdx.x >> 5;

  const int bh = blockIdx.x >> 4;             // b*NHEAD + h   (128 heads)
  const int qt = (blockIdx.x & 15) * 4 + warp;  // 16-query tile in [0,64)
  const int b  = bh >> 4;
  const int h  = bh & 15;

  const _Float16* qb = qh + (size_t)bh * SEQ * HDIM;
  const _Float16* kb = kh + (size_t)bh * SEQ * HDIM;
  const _Float16* vb = vT + (size_t)bh * HDIM * SEQ;
  _Float16* pl = pbase + (size_t)warp * 16 * PSTR;

  // ---- stage 0: one wave DMAs K(head) into LDS via the Tensor Data Mover
  if (warp == 0) tdm_load_k_head(kb, ksh);
  __syncthreads();

  const int half = lane >> 4;
  const int lrow = lane & 15;
  const int q    = qt * 16 + lrow;

  // Q^T as B operand: col = query q, contiguous d
  const v16h qB0 = load_b16(qb + (size_t)q * HDIM + half * 16);       // d 0..31
  const v16h qB1 = load_b16(qb + (size_t)q * HDIM + 32 + half * 16);  // d 32..63

  const float scale = 0.125f;  // HDIM^-0.5

  // ---- pass A: row max (K streamed from LDS) -----------------------------
  float mloc = -3.0e30f;
  for (int kt = 0; kt < SEQ / 16; ++kt) {
    const _Float16* krow = ksh + (size_t)(kt * 16 + lrow) * KROW;
    v16h a0 = load_a16(krow, half);
    v16h a1 = load_a16(krow + 32, half);
    v8f c = {};
    c = wmma_f16(a0, qB0, c);
    c = wmma_f16(a1, qB1, c);
#pragma unroll
    for (int r = 0; r < 8; ++r) mloc = fmaxf(mloc, c[r]);
  }
  float mrow = mloc * scale;
  mrow = fmaxf(mrow, __shfl_xor(mrow, 16, 32));

  // ---- pass B: exp, sum, probs -> LDS ------------------------------------
  float ssum = 0.0f;
  for (int kt = 0; kt < SEQ / 16; ++kt) {
    const _Float16* krow = ksh + (size_t)(kt * 16 + lrow) * KROW;
    v16h a0 = load_a16(krow, half);
    v16h a1 = load_a16(krow + 32, half);
    v8f c = {};
    c = wmma_f16(a0, qB0, c);
    c = wmma_f16(a1, qB1, c);
    v8h pv;
#pragma unroll
    for (int r = 0; r < 8; ++r) {
      const float e = __expf(c[r] * scale - mrow);
      ssum += e;
      pv[r] = (_Float16)e;
    }
    // lane writes keys kt*16 + half*8 + (0..7) of its query row: 16B aligned
    *(v8h*)(pl + (size_t)lrow * PSTR + kt * 16 + half * 8) = pv;
  }
  float srow = ssum + __shfl_xor(ssum, 16, 32);
  const float rs = 1.0f / srow;

  __syncthreads();  // order prob writes vs phase-3 LDS reads (uniform)

  // ---- phase 3: out^T = V^T x P^T ----------------------------------------
#pragma unroll
  for (int dt = 0; dt < 4; ++dt) {
    const _Float16* vrow = vb + (size_t)(dt * 16 + lrow) * SEQ;
    v8f acc = {};
    for (int kc = 0; kc < SEQ / 32; ++kc) {
      __builtin_prefetch(vrow + kc * 32 + 256, 0, 1);
      v16h a = load_a16(vrow + kc * 32, half);                       // V^T rows
      v16h bm = load_b16(pl + (size_t)lrow * PSTR + kc * 32 + half * 16);
      acc = wmma_f16(a, bm, acc);
    }
    // C layout: col = query (lrow), rows d = dt*16 + half*8 + r (consecutive)
    v8h ov;
#pragma unroll
    for (int r = 0; r < 8; ++r) ov[r] = (_Float16)(acc[r] * rs);
    const size_t row = (size_t)(b * SEQ + q);
    *(v8h*)(attn + row * CDIM + h * HDIM + dt * 16 + half * 8) = ov;
  }
}

// ---------------------------------------------------------------------------
// Host-side launch
// ---------------------------------------------------------------------------
extern "C" void kernel_launch(void* const* d_in, const int* in_sizes, int n_in,
                              void* d_out, int out_size, void* d_ws,
                              size_t ws_size, hipStream_t stream) {
  const float* x      = (const float*)d_in[0];   // (8,1024,1024)
  const float* W_qkv  = (const float*)d_in[1];   // (1024,3072)
  const float* b_qkv  = (const float*)d_in[2];   // (3072)
  const float* W_proj = (const float*)d_in[3];   // (1024,1024)
  const float* b_proj = (const float*)d_in[4];   // (1024)
  float* out = (float*)d_out;                    // (8,1024,1024) f32

  const int M = BATCH * SEQ;  // 8192
  char* ws = (char*)d_ws;
  size_t off = 0;
  auto carve = [&](size_t bytes) {
    char* p = ws + off;
    off = (off + bytes + 255) & ~(size_t)255;
    return p;
  };
  _Float16* xh     = (_Float16*)carve((size_t)M * CDIM * 2);
  _Float16* wqkvT  = (_Float16*)carve((size_t)N_QKV * CDIM * 2);
  _Float16* wprojT = (_Float16*)carve((size_t)CDIM * CDIM * 2);
  _Float16* qkv16  = (_Float16*)carve((size_t)M * N_QKV * 2);
  _Float16* qhb    = (_Float16*)carve((size_t)M * CDIM * 2);
  _Float16* khb    = (_Float16*)carve((size_t)M * CDIM * 2);
  _Float16* vTb    = (_Float16*)carve((size_t)M * CDIM * 2);
  _Float16* attn   = (_Float16*)carve((size_t)M * CDIM * 2);
  (void)ws_size;

  // 1) convert + transpose weights/activations to f16
  k_f32_to_f16<<<4096, 256, 0, stream>>>(x, xh, M * CDIM);
  k_transpose_f16<<<4096, 256, 0, stream>>>(W_qkv, wqkvT, CDIM, N_QKV);
  k_transpose_f16<<<2048, 256, 0, stream>>>(W_proj, wprojT, CDIM, CDIM);

  // 2) QKV GEMM: waves = (M/16)*(N_QKV/64) = 512*48 = 24576; 4 waves/block
  k_wmma_gemm_bias<_Float16><<<24576 / 4, 128, 0, stream>>>(
      xh, wqkvT, b_qkv, qkv16, M, N_QKV, CDIM);

  // 3) RoPE + layout shuffle (one thread per (b,h,s,d))
  k_rope_reshape<<<(BATCH * NHEAD * SEQ * HDIM) / 256, 256, 0, stream>>>(
      qkv16, qhb, khb, vTb);

  // 4) attention: 128 heads x 16 blocks, 4 waves/block sharing the TDM-staged
  //    K head; 144KB K + 130KB probs = 274KB of the WGP's 320KB LDS
  const size_t lds_bytes = (size_t)(SEQ * KROW + 4 * 16 * PSTR) * sizeof(_Float16);
  k_attention<<<2048, 128, lds_bytes, stream>>>(qhb, khb, vTb, attn);

  // 5) output projection: waves = 512*(1024/64) = 8192
  k_wmma_gemm_bias<float><<<8192 / 4, 128, 0, stream>>>(
      attn, wprojT, b_proj, out, M, CDIM, CDIM);
}
